// MultiHeadSelfAttention_16114717295314
// MI455X (gfx1250) — matrix operand order repaired and checked by reading, not compiled, not run on hardware
//
#include <hip/hip_runtime.h>
#include <stdint.h>

#define DIM   1024
#define HEADS 16
#define HD    64
#define SEQ   2048
#define BATCH 2

typedef __attribute__((ext_vector_type(16))) __bf16 v16bf;
typedef __attribute__((ext_vector_type(8)))  float  v8f;

union AFrag { v16bf v; uint32_t u[8]; };

// ---- async global->LDS staging (CDNA5 GLOBAL_LOAD_ASYNC_TO_LDS_B128) ----
#if defined(__has_builtin)
#if __has_builtin(__builtin_amdgcn_global_load_async_to_lds_b128) && \
    __has_builtin(__builtin_amdgcn_s_wait_asynccnt)
#define USE_ASYNC_LDS 1
#endif
#endif

typedef int v4i_vs __attribute__((vector_size(16)));
typedef __attribute__((address_space(1))) v4i_vs* gptr_v4i;
typedef __attribute__((address_space(3))) v4i_vs* lptr_v4i;

__device__ __forceinline__ void cp16_to_lds(uint16_t* lds_dst,
                                            const uint16_t* gsrc) {
#ifdef USE_ASYNC_LDS
  __builtin_amdgcn_global_load_async_to_lds_b128(
      (gptr_v4i)(uint16_t*)gsrc, (lptr_v4i)lds_dst, 0, 0);
#else
  *(uint4*)lds_dst = *(const uint4*)gsrc;
#endif
}
__device__ __forceinline__ void async_wait() {
#ifdef USE_ASYNC_LDS
  __builtin_amdgcn_s_wait_asynccnt(0);
#endif
}

// ---- bf16 conversion helpers (native cvt via __bf16 cast, RNE) ----
__device__ __forceinline__ uint32_t pack2bf(float a, float b) {
  union { __bf16 h[2]; uint32_t u; } p;
  p.h[0] = (__bf16)a;
  p.h[1] = (__bf16)b;
  return p.u;
}
__device__ __forceinline__ void st_bf(uint16_t* p, float f) {
  *(__bf16*)p = (__bf16)f;
}

// ---- WMMA fragment gathers (CDNA5 ISA 7.12.2, 16-bit layouts, wave32) ----
// A 16x32 bf16: lanes 0-15: M=lane, VGPR0-3 K=0..7, VGPR4-7 K=16..23
//               lanes16-31: M=lane-16, VGPR0-3 K=8..15, VGPR4-7 K=24..31
__device__ __forceinline__ v16bf load_A_frag(const uint16_t* base, int ld) {
  int l   = threadIdx.x & 31;
  int m   = l & 15;
  int klo = (l < 16) ? 0 : 8;
  const uint16_t* p = base + m * ld;
  AFrag f;
#pragma unroll
  for (int j = 0; j < 4; ++j) {
    f.u[j]     = *(const uint32_t*)(p + klo + 2 * j);        // K = klo+2j, +1
    f.u[j + 4] = *(const uint32_t*)(p + 16 + klo + 2 * j);   // K = 16+klo+2j
  }
  return f.v;
}

// B 32x16 bf16 gathered from a [N][K] (transposed) tile:
// lanes 0-15: N=lane, VGPR0-3 K=0..7, VGPR4-7 K=16..23; lanes 16-31: N=lane-16, VGPR0-3 K=8..15, VGPR4-7 K=24..31
__device__ __forceinline__ v16bf load_B_frag(const uint16_t* base, int ld) {
  int l  = threadIdx.x & 31;
  int n  = l & 15;
  int kb = (l < 16) ? 0 : 8;
  const uint16_t* p = base + n * ld + kb;
  AFrag f;
#pragma unroll
  for (int j = 0; j < 8; ++j) f.u[j] = *(const uint32_t*)(p + 2 * j + (j >= 4 ? 8 : 0));
  return f.v;
}

__device__ __forceinline__ v8f wmma_bf16(v16bf a, v16bf b, v8f c) {
  return __builtin_amdgcn_wmma_f32_16x16x32_bf16(false, a, false, b, (short)0, c,
                                                 false, false);
}

#define ZERO8 {0,0,0,0,0,0,0,0}

// ============  Kernel 0a: X f32 -> bf16 (one-time, memory-bound)  ============
__global__ __launch_bounds__(256) void cvt_x_kernel(const float* __restrict__ X,
                                                    uint16_t* __restrict__ Xb) {
  int i = (blockIdx.x * 256 + threadIdx.x) * 4;
  float4 f = *(const float4*)(X + i);
  *(uint32_t*)(Xb + i)     = pack2bf(f.x, f.y);
  *(uint32_t*)(Xb + i + 2) = pack2bf(f.z, f.w);
}

// ======  Kernel 0b: W[K][N] f32 -> Wt[N][K] bf16 (LDS-tiled transpose)  ======
__global__ __launch_bounds__(256) void cvt_wT_kernel(const float* __restrict__ W,
                                                     uint16_t* __restrict__ Wt,
                                                     int K, int N) {
  __shared__ uint16_t tile[32][33];
  const int tid = threadIdx.x;
  const int kb = blockIdx.y * 32, nb = blockIdx.x * 32;
  {
    int r = tid >> 3, c = (tid & 7) * 4;            // coalesced f32 reads
    float4 f = *(const float4*)(W + (kb + r) * N + nb + c);
    tile[r][c + 0] = (uint16_t)(pack2bf(f.x, 0.f) & 0xFFFFu);
    tile[r][c + 1] = (uint16_t)(pack2bf(f.y, 0.f) & 0xFFFFu);
    tile[r][c + 2] = (uint16_t)(pack2bf(f.z, 0.f) & 0xFFFFu);
    tile[r][c + 3] = (uint16_t)(pack2bf(f.w, 0.f) & 0xFFFFu);
  }
  __syncthreads();
  {
    int n = tid >> 3, k0 = (tid & 7) * 4;           // coalesced bf16 writes
    uint32_t lo = (uint32_t)tile[k0 + 0][n] | ((uint32_t)tile[k0 + 1][n] << 16);
    uint32_t hi = (uint32_t)tile[k0 + 2][n] | ((uint32_t)tile[k0 + 3][n] << 16);
    uint16_t* dst = Wt + (size_t)(nb + n) * K + kb + k0;
    *(uint32_t*)(dst)     = lo;
    *(uint32_t*)(dst + 2) = hi;
  }
}

// =====================  Kernel 1: QKV GEMM + bias  =====================
// C[4096,3072] = Xb[4096,1024] @ WqkvT[3072,1024]^T + b ; scatter bf16 Q/K/V
// WG: 128 thr = 4 waves; tile 128x64; wave tile 32x64 (8 WMMA / K-step)
__global__ __launch_bounds__(128) void qkv_gemm_kernel(
    const uint16_t* __restrict__ Xb, const uint16_t* __restrict__ WqkvT,
    const float* __restrict__ bias, uint16_t* __restrict__ qws,
    uint16_t* __restrict__ kws, uint16_t* __restrict__ vws) {
  __shared__ uint16_t Xs[128 * 32];  // [m][k]
  __shared__ uint16_t Ws[64 * 32];   // [n][k]
  const int tid = threadIdx.x, lane = tid & 31, wave = tid >> 5;
  const int mbase = blockIdx.y * 128, nbase = blockIdx.x * 64;

  v8f acc[2][4] = {{ZERO8, ZERO8, ZERO8, ZERO8}, {ZERO8, ZERO8, ZERO8, ZERO8}};

  for (int kc = 0; kc < DIM; kc += 32) {
    __syncthreads();
    if (kc + 32 < DIM) {
      __builtin_prefetch(Xb + (mbase + tid) * DIM + kc + 32, 0, 1);
      __builtin_prefetch(WqkvT + (size_t)(nbase + (tid >> 1)) * DIM + kc + 32, 0, 1);
    }
#pragma unroll
    for (int i = 0; i < 4; ++i) {                     // Xs: 128x32 bf16 stage
      int e = (tid + i * 128) * 8;
      int r = e >> 5, c = e & 31;
      cp16_to_lds(Xs + e, Xb + (mbase + r) * DIM + kc + c);
    }
#pragma unroll
    for (int i = 0; i < 2; ++i) {                     // Ws: 64x32 bf16 stage
      int e = (tid + i * 128) * 8;
      int n = e >> 5, k = e & 31;
      cp16_to_lds(Ws + e, WqkvT + (size_t)(nbase + n) * DIM + kc + k);
    }
    async_wait();
    __syncthreads();
    v16bf a0 = load_A_frag(Xs + (wave * 32) * 32, 32);
    v16bf a1 = load_A_frag(Xs + (wave * 32 + 16) * 32, 32);
    v16bf b[4];
#pragma unroll
    for (int t = 0; t < 4; ++t) b[t] = load_B_frag(Ws + t * 16 * 32, 32);
#pragma unroll
    for (int t = 0; t < 4; ++t) acc[0][t] = wmma_bf16(a0, b[t], acc[0][t]);
#pragma unroll
    for (int t = 0; t < 4; ++t) acc[1][t] = wmma_bf16(a1, b[t], acc[1][t]);
  }

  // Epilogue: bias, split into Q/K/V (col: part=n/1024, h=(n%1024)/64, d=n%64)
  const int half = lane >> 4;
#pragma unroll
  for (int st = 0; st < 2; ++st) {
    const int mrow = mbase + wave * 32 + st * 16 + half * 8;
#pragma unroll
    for (int t = 0; t < 4; ++t) {
      int n = nbase + t * 16 + (lane & 15);
      float bv = bias[n];
      int part = n >> 10, rem = n & 1023;
      int h = rem >> 6, d = rem & 63;
      uint16_t* dst = (part == 0) ? qws : (part == 1) ? kws : vws;
#pragma unroll
      for (int r = 0; r < 8; ++r) {
        int row = mrow + r;
        int b2 = row >> 11, s = row & (SEQ - 1);
        st_bf(dst + (((b2 * HEADS + h) * SEQ) + s) * HD + d, acc[st][t][r] + bv);
      }
    }
  }
}

// =================  Kernel 2: flash attention (per b,h)  =================
__global__ __launch_bounds__(256) void attn_kernel(
    const uint16_t* __restrict__ qws, const uint16_t* __restrict__ kws,
    const uint16_t* __restrict__ vws, uint16_t* __restrict__ aout) {
  __shared__ uint16_t Ks[64 * 64];         // [key][d]   (B-layout for scores)
  __shared__ uint16_t Vt[64 * 64];         // [d][key]   (B-layout for P·V)
  __shared__ uint16_t Ps[8 * 16 * 64];     // per-wave P tile [16][64]

  const int tid = threadIdx.x, lane = tid & 31, wave = tid >> 5;
  const int half = lane >> 4, n16 = lane & 15;
  const int bh = blockIdx.y;               // 0..31
  const int b = bh >> 4, h = bh & 15;
  const int q0 = blockIdx.x * 128 + wave * 16;

  const uint16_t* qbase = qws + ((size_t)bh * SEQ + q0) * HD;
  const uint16_t* kbase = kws + (size_t)bh * SEQ * HD;
  const uint16_t* vbase = vws + (size_t)bh * SEQ * HD;

  v16bf aq0 = load_A_frag(qbase, HD);
  v16bf aq1 = load_A_frag(qbase + 32, HD);

  v8f o[4] = {ZERO8, ZERO8, ZERO8, ZERO8};
  float run_m[8], run_l[8];
#pragma unroll
  for (int r = 0; r < 8; ++r) { run_m[r] = -1e30f; run_l[r] = 0.0f; }
  const float sc = 0.125f;  // 1/sqrt(64)
  uint16_t* Pw = Ps + wave * 16 * 64;

  for (int kc = 0; kc < SEQ; kc += 64) {
    __syncthreads();
#pragma unroll
    for (int i = 0; i < 2; ++i) {          // stage K (async) and V (transposed)
      int e = (tid + i * 256) * 8;         // chunk is contiguous 4096 elems
      cp16_to_lds(Ks + e, kbase + kc * HD + e);
      union { uint4 q; uint16_t hh[8]; } v;
      v.q = *(const uint4*)(vbase + kc * HD + e);
      int row = e >> 6, c0 = e & 63;
#pragma unroll
      for (int j = 0; j < 8; ++j) Vt[(c0 + j) * 64 + row] = v.hh[j];
    }
    async_wait();
    __syncthreads();

    // ---- scores: S(16x64) = Q(16x64) . K^T ----
    v8f s[4] = {ZERO8, ZERO8, ZERO8, ZERO8};
    {
      v16bf bk[4];
#pragma unroll
      for (int t = 0; t < 4; ++t) bk[t] = load_B_frag(Ks + t * 16 * 64, 64);
#pragma unroll
      for (int t = 0; t < 4; ++t) s[t] = wmma_bf16(aq0, bk[t], s[t]);
#pragma unroll
      for (int t = 0; t < 4; ++t) bk[t] = load_B_frag(Ks + t * 16 * 64 + 32, 64);
#pragma unroll
      for (int t = 0; t < 4; ++t) s[t] = wmma_bf16(aq1, bk[t], s[t]);
    }
#pragma unroll
    for (int t = 0; t < 4; ++t)
#pragma unroll
      for (int r = 0; r < 8; ++r) s[t][r] *= sc;

    // ---- online softmax: row r lives in one 16-lane half, VGPR r ----
    float bm[8];
#pragma unroll
    for (int r = 0; r < 8; ++r) {
      bm[r] = fmaxf(fmaxf(s[0][r], s[1][r]), fmaxf(s[2][r], s[3][r]));
#pragma unroll
      for (int mask = 1; mask <= 8; mask <<= 1)
        bm[r] = fmaxf(bm[r], __shfl_xor(bm[r], mask, 32));
    }
    float corr[8], rs[8];
#pragma unroll
    for (int r = 0; r < 8; ++r) {
      float nm = fmaxf(run_m[r], bm[r]);
      corr[r] = __expf(run_m[r] - nm);
      run_m[r] = nm;
      rs[r] = 0.0f;
    }
#pragma unroll
    for (int t = 0; t < 4; ++t)
#pragma unroll
      for (int r = 0; r < 8; ++r) {
        float p = __expf(s[t][r] - run_m[r]);
        rs[r] += p;
        st_bf(Pw + (r + half * 8) * 64 + t * 16 + n16, p);  // re-layout via LDS
      }
#pragma unroll
    for (int r = 0; r < 8; ++r) {
#pragma unroll
      for (int mask = 1; mask <= 8; mask <<= 1)
        rs[r] += __shfl_xor(rs[r], mask, 32);
      run_l[r] = run_l[r] * corr[r] + rs[r];
    }
#pragma unroll
    for (int t = 0; t < 4; ++t)
#pragma unroll
      for (int r = 0; r < 8; ++r) o[t][r] *= corr[r];

    __syncthreads();
    // ---- O += P(16x64) . V(64x64) ----
#pragma unroll
    for (int c = 0; c < 2; ++c) {
      v16bf ap = load_A_frag(Pw + c * 32, 64);
      v16bf bv[4];
#pragma unroll
      for (int t = 0; t < 4; ++t)
        bv[t] = load_B_frag(Vt + t * 16 * 64 + c * 32, 64);
#pragma unroll
      for (int t = 0; t < 4; ++t) o[t] = wmma_bf16(ap, bv[t], o[t]);
    }
  }

  // Epilogue: normalize and write bf16 attn output [B,S,D] (D = h*64+d)
  float inv[8];
#pragma unroll
  for (int r = 0; r < 8; ++r) inv[r] = 1.0f / run_l[r];
#pragma unroll
  for (int t = 0; t < 4; ++t)
#pragma unroll
    for (int r = 0; r < 8; ++r) {
      int srow = q0 + r + half * 8;
      st_bf(aout + ((size_t)b * SEQ + srow) * DIM + h * HD + t * 16 + n16,
            o[t][r] * inv[r]);
    }
}

// ====================  Kernel 3: projection GEMM  ====================
// out[4096,1024] = A(bf16)[4096,1024] @ WprojT[1024,1024]^T + b (f32 out)
__global__ __launch_bounds__(128) void proj_gemm_kernel(
    const uint16_t* __restrict__ A, const uint16_t* __restrict__ WprojT,
    const float* __restrict__ bias, float* __restrict__ out) {
  __shared__ uint16_t As[128 * 32];
  __shared__ uint16_t Ws[64 * 32];
  const int tid = threadIdx.x, lane = tid & 31, wave = tid >> 5;
  const int mbase = blockIdx.y * 128, nbase = blockIdx.x * 64;

  v8f acc[2][4] = {{ZERO8, ZERO8, ZERO8, ZERO8}, {ZERO8, ZERO8, ZERO8, ZERO8}};

  for (int kc = 0; kc < DIM; kc += 32) {
    __syncthreads();
    if (kc + 32 < DIM) {
      __builtin_prefetch(A + (mbase + tid) * DIM + kc + 32, 0, 1);
      __builtin_prefetch(WprojT + (size_t)(nbase + (tid >> 1)) * DIM + kc + 32, 0, 1);
    }
#pragma unroll
    for (int i = 0; i < 4; ++i) {
      int e = (tid + i * 128) * 8;
      int r = e >> 5, c = e & 31;
      cp16_to_lds(As + e, A + (mbase + r) * DIM + kc + c);
    }
#pragma unroll
    for (int i = 0; i < 2; ++i) {
      int e = (tid + i * 128) * 8;
      int n = e >> 5, k = e & 31;
      cp16_to_lds(Ws + e, WprojT + (size_t)(nbase + n) * DIM + kc + k);
    }
    async_wait();
    __syncthreads();
    v16bf a0 = load_A_frag(As + (wave * 32) * 32, 32);
    v16bf a1 = load_A_frag(As + (wave * 32 + 16) * 32, 32);
    v16bf b[4];
#pragma unroll
    for (int t = 0; t < 4; ++t) b[t] = load_B_frag(Ws + t * 16 * 32, 32);
#pragma unroll
    for (int t = 0; t < 4; ++t) acc[0][t] = wmma_bf16(a0, b[t], acc[0][t]);
#pragma unroll
    for (int t = 0; t < 4; ++t) acc[1][t] = wmma_bf16(a1, b[t], acc[1][t]);
  }

  const int half = lane >> 4;
#pragma unroll
  for (int st = 0; st < 2; ++st) {
    const int mrow = mbase + wave * 32 + st * 16 + half * 8;
#pragma unroll
    for (int t = 0; t < 4; ++t) {
      int n = nbase + t * 16 + (lane & 15);
      float bv = bias[n];
#pragma unroll
      for (int r = 0; r < 8; ++r)
        out[(size_t)(mrow + r) * DIM + n] = acc[st][t][r] + bv;
    }
  }
}

// ============================  launcher  ============================
extern "C" void kernel_launch(void* const* d_in, const int* in_sizes, int n_in,
                              void* d_out, int out_size, void* d_ws, size_t ws_size,
                              hipStream_t stream) {
  const float* x     = (const float*)d_in[0];   // [2,2048,1024]
  const float* Wqkv  = (const float*)d_in[1];   // [1024,3072]
  const float* bqkv  = (const float*)d_in[2];   // [3072]
  const float* Wproj = (const float*)d_in[3];   // [1024,1024]
  const float* bproj = (const float*)d_in[4];   // [1024]
  float* out = (float*)d_out;                   // [2,2048,1024] f32

  const size_t NTOK = (size_t)BATCH * SEQ * DIM;  // 4M elems
  uint16_t* xb   = (uint16_t*)d_ws;               // bf16 X          (4M)
  uint16_t* wqt  = xb + NTOK;                     // bf16 Wqkv^T     (3M)
  uint16_t* wpt  = wqt + (size_t)DIM * 3 * DIM;   // bf16 Wproj^T    (1M)
  uint16_t* q_ws = wpt + (size_t)DIM * DIM;       // bf16 Q [B,H,S,64]
  uint16_t* k_ws = q_ws + NTOK;
  uint16_t* v_ws = k_ws + NTOK;
  uint16_t* a_ws = v_ws + NTOK;                   // bf16 attn [B,S,D]

  // One-time precision conversion (memory-bound, ~3 us at 23.3 TB/s)
  cvt_x_kernel<<<NTOK / 1024, 256, 0, stream>>>(x, xb);
  cvt_wT_kernel<<<dim3(3 * DIM / 32, DIM / 32), 256, 0, stream>>>(Wqkv, wqt,
                                                                  DIM, 3 * DIM);
  cvt_wT_kernel<<<dim3(DIM / 32, DIM / 32), 256, 0, stream>>>(Wproj, wpt,
                                                              DIM, DIM);
  // QKV: 128x64 tiles over [4096, 3072]
  qkv_gemm_kernel<<<dim3(48, 32), 128, 0, stream>>>(xb, wqt, bqkv,
                                                    q_ws, k_ws, v_ws);
  // Attention: 128 query rows per WG, one (b,h) per blockIdx.y
  attn_kernel<<<dim3(SEQ / 128, BATCH * HEADS), 256, 0, stream>>>(
      q_ws, k_ws, v_ws, a_ws);
  // Projection: 128x64 tiles over [4096, 1024]
  proj_gemm_kernel<<<dim3(16, 32), 128, 0, stream>>>(a_ws, wpt, bproj, out);
}
